// MultiHeadAttention_68135361184156
// MI455X (gfx1250) — matrix operand order repaired and checked by reading, not compiled, not run on hardware
//
#include <hip/hip_runtime.h>
#include <hip/hip_bf16.h>

// ---------------------------------------------------------------------------
// MI455X (gfx1250) multi-head attention with Shaw relative positions.
// All GEMMs via v_wmma_f32_16x16x32_bf16 (wave32, f32 accumulate).
// Each wave computes a 32(M) x 64(N) tile: 2 A-tiles share 4 B-tiles,
// 8 WMMAs per K-step. global_prefetch_b8 warms the next K-step.
// ---------------------------------------------------------------------------

typedef __attribute__((ext_vector_type(16))) __bf16 v16bf;
typedef __attribute__((ext_vector_type(8)))  float  v8f;

struct BF16x16 { uint4 lo, hi; };   // 32 bytes == one v16bf operand

__device__ __forceinline__ unsigned short f2bfu(float f) {
    unsigned u = __builtin_bit_cast(unsigned, f);
    u += 0x7FFFu + ((u >> 16) & 1u);          // round-to-nearest-even
    return (unsigned short)(u >> 16);
}
__device__ __forceinline__ unsigned pack2(float a, float b) {
    return (unsigned)f2bfu(a) | ((unsigned)f2bfu(b) << 16);
}

__device__ __forceinline__ v8f wmma_bf16(v16bf a, v16bf b, v8f c) {
    // 8 args: (neg_a, A, neg_b, B, c_mod, C, reuse_a, reuse_b)
    return __builtin_amdgcn_wmma_f32_16x16x32_bf16(false, a, false, b,
                                                   (short)0, c, false, false);
}

// A-tile (16x32, bf16) from row-major source (contiguous K, matches B):
// lane<16: row=lane,    elems 0-15 = K 0-15
// lane>=16: row=lane-16, elems 0-15 = K 16-31
__device__ __forceinline__ v16bf load_a_bf16(const unsigned short* base, int ld,
                                             int m0, int k0, int lane) {
    const int row = m0 + (lane & 15);
    const int ko  = k0 + ((lane >> 4) << 4);
    const unsigned short* p = base + (size_t)row * ld + ko;
    BF16x16 r;
    r.lo = *reinterpret_cast<const uint4*>(p);
    r.hi = *reinterpret_cast<const uint4*>(p + 8);
    return __builtin_bit_cast(v16bf, r);
}

// Same A-tile but fp32 source, converted to bf16 on the fly.
__device__ __forceinline__ v16bf load_a_f32(const float* base, int ld,
                                            int m0, int k0, int lane) {
    const int row = m0 + (lane & 15);
    const int ko  = k0 + ((lane >> 4) << 4);
    const float* p = base + (size_t)row * ld + ko;
    float4 f0 = *reinterpret_cast<const float4*>(p);
    float4 f1 = *reinterpret_cast<const float4*>(p + 4);
    float4 f2 = *reinterpret_cast<const float4*>(p + 8);
    float4 f3 = *reinterpret_cast<const float4*>(p + 12);
    BF16x16 r;
    r.lo.x = pack2(f0.x, f0.y); r.lo.y = pack2(f0.z, f0.w);
    r.lo.z = pack2(f1.x, f1.y); r.lo.w = pack2(f1.z, f1.w);
    r.hi.x = pack2(f2.x, f2.y); r.hi.y = pack2(f2.z, f2.w);
    r.hi.z = pack2(f3.x, f3.y); r.hi.w = pack2(f3.z, f3.w);
    return __builtin_bit_cast(v16bf, r);
}

// B-tile (32x16 KxN, bf16). Source is the NxK matrix row-major (row = output
// column n). lanes 0-15: col=lane, K k0..k0+15; lanes 16-31: col=lane-16,
// K k0+16..k0+31 (contiguous per lane -> two 16B loads).
__device__ __forceinline__ v16bf load_b_bf16(const unsigned short* base, int ld,
                                             int n0, int k0, int lane) {
    const int col = n0 + (lane & 15);
    const int ko  = k0 + ((lane >> 4) << 4);
    const unsigned short* p = base + (size_t)col * ld + ko;
    BF16x16 r;
    r.lo = *reinterpret_cast<const uint4*>(p);
    r.hi = *reinterpret_cast<const uint4*>(p + 8);
    return __builtin_bit_cast(v16bf, r);
}

// ---------------------------------------------------------------------------
// fp32 -> bf16 conversion
// ---------------------------------------------------------------------------
__global__ void cvt_f32_bf16_kernel(const float* __restrict__ src,
                                    unsigned short* __restrict__ dst, int n) {
    int i = blockIdx.x * blockDim.x + threadIdx.x;
    if (i < n) dst[i] = f2bfu(src[i]);
}

// rel_v_table (65x64) -> transposed + zero-padded bf16 [64][96]
__global__ void build_rel_vt_kernel(const float* __restrict__ rel_v,
                                    unsigned short* __restrict__ rel_vt) {
    int i = blockIdx.x * blockDim.x + threadIdx.x;
    if (i >= 64 * 96) return;
    int d = i / 96, r = i % 96;
    float v = (r < 65) ? rel_v[r * 64 + d] : 0.f;
    rel_vt[d * 96 + r] = f2bfu(v);
}

// ---------------------------------------------------------------------------
// Y[m,n] = sum_k A[m,k] * W[n,k] + bias[n].  One wave -> 32(M) x 64(N) tile.
// mode 0: store bf16 as [b][h][s][d]   (Q/K heads layout)
// mode 1: store bf16 as [b][h][d][s]   (V transposed for attn@V B-operand)
// mode 2: store fp32 as [m][n]         (final output projection)
// ---------------------------------------------------------------------------
__global__ void gemm_xwT_kernel(const unsigned short* __restrict__ A,
                                const unsigned short* __restrict__ W,
                                const float* __restrict__ bias,
                                void* __restrict__ dst,
                                int M, int N, int K, int mode) {
    const int wave = (int)((blockIdx.x * blockDim.x + threadIdx.x) >> 5);
    const int lane = threadIdx.x & 31;
    const int mtiles = M >> 5;                 // 32-row tiles
    const int mtile  = wave % mtiles;
    const int nstrip = wave / mtiles;
    const int m0 = mtile << 5;
    const int n0 = nstrip << 6;
    if (n0 >= N) return;

    v8f c[2][4] = {};
    for (int k0 = 0; k0 < K; k0 += 32) {
        // prefetch next K-step (speculative; dropped if OOB)
        __builtin_prefetch(A + (size_t)(m0 + (lane & 15)) * K + k0 + 32, 0, 1);
        __builtin_prefetch(W + (size_t)(n0 + (lane & 15)) * K + k0 + 32, 0, 1);
        v16bf a0 = load_a_bf16(A, K, m0,      k0, lane);
        v16bf a1 = load_a_bf16(A, K, m0 + 16, k0, lane);
        #pragma unroll
        for (int j = 0; j < 4; ++j) {
            v16bf b = load_b_bf16(W, K, n0 + j * 16, k0, lane);
            c[0][j] = wmma_bf16(a0, b, c[0][j]);
            c[1][j] = wmma_bf16(a1, b, c[1][j]);
        }
    }
    const int nl = lane & 15;
    #pragma unroll
    for (int mt = 0; mt < 2; ++mt) {
        const int mb = m0 + mt * 16 + ((lane >> 4) << 3);
        #pragma unroll
        for (int j = 0; j < 4; ++j) {
            const int n = n0 + j * 16 + nl;
            const float bn = bias[n];
            #pragma unroll
            for (int i = 0; i < 8; ++i) {
                const int m = mb + i;
                const float val = c[mt][j][i] + bn;
                if (mode == 0) {
                    const int b_ = m >> 10, s_ = m & 1023, h_ = n >> 6, d_ = n & 63;
                    ((unsigned short*)dst)[(((size_t)b_ * 16 + h_) * 1024 + s_) * 64 + d_] = f2bfu(val);
                } else if (mode == 1) {
                    const int b_ = m >> 10, s_ = m & 1023, h_ = n >> 6, d_ = n & 63;
                    ((unsigned short*)dst)[(((size_t)b_ * 16 + h_) * 64 + d_) * 1024 + s_] = f2bfu(val);
                } else {
                    ((float*)dst)[(size_t)m * N + n] = val;
                }
            }
        }
    }
}

// ---------------------------------------------------------------------------
// Rq[m][r] = sum_d Qb[m][d] * rel_k[r][d],  m in [0,65536), r in [0,65).
// One wave per 16-row tile; 5 N-tiles (r padded to 80, clamped loads).
// ---------------------------------------------------------------------------
__global__ void rq_gemm_kernel(const unsigned short* __restrict__ Qb,
                               const unsigned short* __restrict__ relk,
                               float* __restrict__ Rq, int Mrows) {
    const int wave = (int)((blockIdx.x * blockDim.x + threadIdx.x) >> 5);
    const int lane = threadIdx.x & 31;
    const int m0 = wave << 4;
    if (m0 >= Mrows) return;

    v8f c[5] = {};
    #pragma unroll
    for (int k0 = 0; k0 < 64; k0 += 32) {
        v16bf a = load_a_bf16(Qb, 64, m0, k0, lane);
        #pragma unroll
        for (int j = 0; j < 5; ++j) {
            int r  = j * 16 + (lane & 15);
            int rc = r > 64 ? 64 : r;   // clamp pad columns (results discarded)
            const unsigned short* p = relk + (size_t)rc * 64 + k0 + ((lane >> 4) << 4);
            BF16x16 raw;
            raw.lo = *reinterpret_cast<const uint4*>(p);
            raw.hi = *reinterpret_cast<const uint4*>(p + 8);
            c[j] = wmma_bf16(a, __builtin_bit_cast(v16bf, raw), c[j]);
        }
    }
    const int nl = lane & 15;
    const int mb = m0 + ((lane >> 4) << 3);
    #pragma unroll
    for (int j = 0; j < 5; ++j) {
        const int r = j * 16 + nl;
        if (r < 65) {
            #pragma unroll
            for (int i = 0; i < 8; ++i)
                Rq[(size_t)(mb + i) * 65 + r] = c[j][i];
        }
    }
}

// ---------------------------------------------------------------------------
// scores[b,h,q,k] = (Q.K^T)/8 + Rq[b,h,q, clip(k-q,+-32)+32]
// One wave -> 32(q) x 64(k) tile. 32768 waves.
// ---------------------------------------------------------------------------
__global__ void scores_kernel(const unsigned short* __restrict__ Qb,
                              const unsigned short* __restrict__ Kb,
                              const float* __restrict__ Rq,
                              float* __restrict__ attn) {
    const int wave = (int)((blockIdx.x * blockDim.x + threadIdx.x) >> 5);
    const int lane = threadIdx.x & 31;
    const int nstrip = wave & 15;
    const int mtile  = (wave >> 4) & 31;
    const int head   = wave >> 9;
    const unsigned short* Qh = Qb + (size_t)head * 1024 * 64;
    const unsigned short* Kh = Kb + (size_t)head * 1024 * 64;
    const int m0 = mtile << 5;
    const int n0 = nstrip << 6;

    v8f c[2][4] = {};
    #pragma unroll
    for (int k0 = 0; k0 < 64; k0 += 32) {
        v16bf a0 = load_a_bf16(Qh, 64, m0,      k0, lane);
        v16bf a1 = load_a_bf16(Qh, 64, m0 + 16, k0, lane);
        #pragma unroll
        for (int j = 0; j < 4; ++j) {
            v16bf b = load_b_bf16(Kh, 64, n0 + j * 16, k0, lane);
            c[0][j] = wmma_bf16(a0, b, c[0][j]);
            c[1][j] = wmma_bf16(a1, b, c[1][j]);
        }
    }
    const int nl = lane & 15;
    const float* RqH   = Rq   + (size_t)head * 1024 * 65;
    float*       attnH = attn + (size_t)head * 1024 * 1024;
    #pragma unroll
    for (int mt = 0; mt < 2; ++mt) {
        const int mb = m0 + mt * 16 + ((lane >> 4) << 3);
        #pragma unroll
        for (int j = 0; j < 4; ++j) {
            const int kk = n0 + j * 16 + nl;
            #pragma unroll
            for (int i = 0; i < 8; ++i) {
                const int q = mb + i;
                int rel = kk - q;
                rel = rel < -32 ? -32 : (rel > 32 ? 32 : rel);
                const float val = c[mt][j][i] * 0.125f + RqH[(size_t)q * 65 + (rel + 32)];
                attnH[(size_t)q * 1024 + kk] = val;
            }
        }
    }
}

// ---------------------------------------------------------------------------
// Row softmax (in place, 1024 wide) + relative-position bucket sums p[96].
// p[r=0] = sum_{k<=q-32} attn;  p[r=64] = sum_{k>=q+32} attn;
// p[1..63] = attn[q, q+r-32] (0 if out of range);  p[65..95] = 0 (padding).
// One 256-thread block per row.
// ---------------------------------------------------------------------------
__global__ void softmax_p_kernel(float* __restrict__ attn, float* __restrict__ p) {
    __shared__ float buf[1024];
    __shared__ float red[256];
    const int t = threadIdx.x;
    const size_t row = blockIdx.x;
    const int q = (int)(row & 1023);
    float* arow = attn + row * 1024;

    float4 v = *(reinterpret_cast<float4*>(arow) + t);
    red[t] = fmaxf(fmaxf(v.x, v.y), fmaxf(v.z, v.w));
    __syncthreads();
    for (int s = 128; s > 0; s >>= 1) {
        if (t < s) red[t] = fmaxf(red[t], red[t + s]);
        __syncthreads();
    }
    const float rmax = red[0];
    __syncthreads();

    float4 e;
    e.x = __expf(v.x - rmax); e.y = __expf(v.y - rmax);
    e.z = __expf(v.z - rmax); e.w = __expf(v.w - rmax);
    red[t] = e.x + e.y + e.z + e.w;
    __syncthreads();
    for (int s = 128; s > 0; s >>= 1) {
        if (t < s) red[t] += red[t + s];
        __syncthreads();
    }
    const float inv = 1.0f / red[0];
    __syncthreads();

    e.x *= inv; e.y *= inv; e.z *= inv; e.w *= inv;
    reinterpret_cast<float4*>(buf)[t] = e;
    *(reinterpret_cast<float4*>(arow) + t) = e;
    __syncthreads();

    float s0 = 0.f, s64 = 0.f;
    for (int k = t; k < 1024; k += 256) {
        const float av = buf[k];
        if (k <= q - 32) s0  += av;
        if (k >= q + 32) s64 += av;
    }
    red[t] = s0; __syncthreads();
    for (int s = 128; s > 0; s >>= 1) { if (t < s) red[t] += red[t + s]; __syncthreads(); }
    const float p0 = red[0]; __syncthreads();
    red[t] = s64; __syncthreads();
    for (int s = 128; s > 0; s >>= 1) { if (t < s) red[t] += red[t + s]; __syncthreads(); }
    const float p64 = red[0];

    float* prow = p + row * 96;
    if (t == 0)            { prow[0] = p0; prow[64] = p64; }
    else if (t < 64)       { int k = q + t - 32; prow[t] = (k >= 0 && k < 1024) ? buf[k] : 0.f; }
    else if (t >= 65 && t < 96) { prow[t] = 0.f; }
}

// ---------------------------------------------------------------------------
// ctx[b,h,q,d] = attn @ V  (K=1024)  +  p @ rel_v^T  (K=96, zero padded).
// One wave -> 32(q) x 64(d). attn/p converted fp32->bf16 in registers.
// Stores ctx bf16 as [b][s][h*64+d] for the output projection.
// ---------------------------------------------------------------------------
__global__ void ctx_kernel(const float* __restrict__ attn,
                           const unsigned short* __restrict__ Vt,
                           const float* __restrict__ p,
                           const unsigned short* __restrict__ rel_vt,
                           unsigned short* __restrict__ ctx) {
    const int wave = (int)((blockIdx.x * blockDim.x + threadIdx.x) >> 5);
    const int lane = threadIdx.x & 31;
    const int head  = wave >> 5;
    const int mtile = wave & 31;
    const int m0 = mtile << 5;
    const float*          Ah = attn + (size_t)head * 1024 * 1024;
    const unsigned short* Vh = Vt   + (size_t)head * 64 * 1024;
    const float*          Ph = p    + (size_t)head * 1024 * 96;

    v8f c[2][4] = {};
    for (int k0 = 0; k0 < 1024; k0 += 32) {
        __builtin_prefetch(Ah + (size_t)(m0 + (lane & 15)) * 1024 + k0 + 32, 0, 1);
        v16bf a0 = load_a_f32(Ah, 1024, m0,      k0, lane);
        v16bf a1 = load_a_f32(Ah, 1024, m0 + 16, k0, lane);
        #pragma unroll
        for (int j = 0; j < 4; ++j) {
            v16bf b = load_b_bf16(Vh, 1024, j * 16, k0, lane);
            c[0][j] = wmma_bf16(a0, b, c[0][j]);
            c[1][j] = wmma_bf16(a1, b, c[1][j]);
        }
    }
    #pragma unroll
    for (int k0 = 0; k0 < 96; k0 += 32) {
        v16bf a0 = load_a_f32(Ph, 96, m0,      k0, lane);
        v16bf a1 = load_a_f32(Ph, 96, m0 + 16, k0, lane);
        #pragma unroll
        for (int j = 0; j < 4; ++j) {
            v16bf b = load_b_bf16(rel_vt, 96, j * 16, k0, lane);
            c[0][j] = wmma_bf16(a0, b, c[0][j]);
            c[1][j] = wmma_bf16(a1, b, c[1][j]);
        }
    }
    const int b_ = head >> 4, hl = head & 15;
    const int nl = lane & 15;
    #pragma unroll
    for (int mt = 0; mt < 2; ++mt) {
        const int mb = m0 + mt * 16 + ((lane >> 4) << 3);
        #pragma unroll
        for (int j = 0; j < 4; ++j) {
            const int d = j * 16 + nl;
            #pragma unroll
            for (int i = 0; i < 8; ++i) {
                const int s = mb + i;
                ctx[((size_t)b_ * 1024 + s) * 1024 + hl * 64 + d] = f2bfu(c[mt][j][i]);
            }
        }
    }
}

// ---------------------------------------------------------------------------
extern "C" void kernel_launch(void* const* d_in, const int* in_sizes, int n_in,
                              void* d_out, int out_size, void* d_ws, size_t ws_size,
                              hipStream_t stream) {
    const float* query = (const float*)d_in[0];
    const float* key_  = (const float*)d_in[1];
    const float* value = (const float*)d_in[2];
    const float* W_q = (const float*)d_in[3];
    const float* b_q = (const float*)d_in[4];
    const float* W_k = (const float*)d_in[5];
    const float* b_k = (const float*)d_in[6];
    const float* W_v = (const float*)d_in[7];
    const float* b_v = (const float*)d_in[8];
    const float* W_o = (const float*)d_in[9];
    const float* b_o = (const float*)d_in[10];
    const float* rel_k = (const float*)d_in[11];
    const float* rel_v = (const float*)d_in[12];

    float* out  = (float*)d_out;
    float* attn = out + (size_t)4 * 1024 * 1024;   // second tuple output

    const size_t M = 4096, D = 1024;               // M = B*S
    char* ws = (char*)d_ws;
    size_t off = 0;
    auto take = [&](size_t bytes) -> char* {
        char* pp = ws + off;
        off += (bytes + 255) & ~(size_t)255;
        return pp;
    };
    unsigned short* xq_bf   = (unsigned short*)take(M * D * 2);
    unsigned short* xk_bf   = (unsigned short*)take(M * D * 2);
    unsigned short* xv_bf   = (unsigned short*)take(M * D * 2);
    unsigned short* wq_bf   = (unsigned short*)take(D * D * 2);
    unsigned short* wk_bf   = (unsigned short*)take(D * D * 2);
    unsigned short* wv_bf   = (unsigned short*)take(D * D * 2);
    unsigned short* wo_bf   = (unsigned short*)take(D * D * 2);
    unsigned short* relk_bf = (unsigned short*)take(65 * 64 * 2);
    unsigned short* relvt_bf= (unsigned short*)take(64 * 96 * 2);
    unsigned short* q_bhsd  = (unsigned short*)take(M * D * 2);
    unsigned short* k_bhsd  = (unsigned short*)take(M * D * 2);
    unsigned short* v_bhds  = (unsigned short*)take(M * D * 2);
    float*          Rq      = (float*)take((size_t)65536 * 65 * 4);
    float*          pbuf    = (float*)take((size_t)65536 * 96 * 4);
    unsigned short* ctx_bf  = (unsigned short*)take(M * D * 2);

    const int T = 256;
    // 1) conversions
    cvt_f32_bf16_kernel<<<(int)((M * D + T - 1) / T), T, 0, stream>>>(query, xq_bf, (int)(M * D));
    cvt_f32_bf16_kernel<<<(int)((M * D + T - 1) / T), T, 0, stream>>>(key_,  xk_bf, (int)(M * D));
    cvt_f32_bf16_kernel<<<(int)((M * D + T - 1) / T), T, 0, stream>>>(value, xv_bf, (int)(M * D));
    cvt_f32_bf16_kernel<<<(int)((D * D + T - 1) / T), T, 0, stream>>>(W_q, wq_bf, (int)(D * D));
    cvt_f32_bf16_kernel<<<(int)((D * D + T - 1) / T), T, 0, stream>>>(W_k, wk_bf, (int)(D * D));
    cvt_f32_bf16_kernel<<<(int)((D * D + T - 1) / T), T, 0, stream>>>(W_v, wv_bf, (int)(D * D));
    cvt_f32_bf16_kernel<<<(int)((D * D + T - 1) / T), T, 0, stream>>>(W_o, wo_bf, (int)(D * D));
    cvt_f32_bf16_kernel<<<(65 * 64 + T - 1) / T, T, 0, stream>>>(rel_k, relk_bf, 65 * 64);
    build_rel_vt_kernel<<<(64 * 96 + T - 1) / T, T, 0, stream>>>(rel_v, relvt_bf);

    // 2) projections: 2048 waves each (128 mtiles x 16 nstrips) = 256 blocks
    gemm_xwT_kernel<<<256, T, 0, stream>>>(xq_bf, wq_bf, b_q, q_bhsd, 4096, 1024, 1024, 0);
    gemm_xwT_kernel<<<256, T, 0, stream>>>(xk_bf, wk_bf, b_k, k_bhsd, 4096, 1024, 1024, 0);
    gemm_xwT_kernel<<<256, T, 0, stream>>>(xv_bf, wv_bf, b_v, v_bhds, 4096, 1024, 1024, 1);

    // 3) Rq = Q . rel_k^T : 4096 waves
    rq_gemm_kernel<<<512, T, 0, stream>>>(q_bhsd, relk_bf, Rq, 65536);

    // 4) scores (+rel gather) into the attn output region: 32768 waves
    scores_kernel<<<4096, T, 0, stream>>>(q_bhsd, k_bhsd, Rq, attn);

    // 5) softmax in place + bucket sums p
    softmax_p_kernel<<<65536, T, 0, stream>>>(attn, pbuf);

    // 6) ctx = attn@V + p@rel_v^T : 2048 waves
    ctx_kernel<<<256, T, 0, stream>>>(attn, v_bhds, pbuf, relvt_bf, ctx_bf);

    // 7) out = ctx @ W_o^T + b_o (fp32 into d_out)
    gemm_xwT_kernel<<<256, T, 0, stream>>>(ctx_bf, wo_bf, b_o, out, 4096, 1024, 1024, 2);
}
